// MultiHeadAttention_2035814499007
// MI455X (gfx1250) — matrix operand order repaired and checked by reading, not compiled, not run on hardware
//
#include <hip/hip_runtime.h>

typedef __bf16 bf16;
typedef __attribute__((ext_vector_type(16))) __bf16 v16bf;
typedef __attribute__((ext_vector_type(8)))  __bf16 v8bf;
typedef __attribute__((ext_vector_type(8)))  float  v8f;
typedef __attribute__((ext_vector_type(4)))  float  v4f;

// ---------------------------------------------------------------- constants
#define BATCH   2
#define SEQ     2048
#define DMODEL  1024
#define NHEADS  16
#define DK      64
#define BL      (BATCH*SEQ)          // 4096

// ---------------------------------------------------------------- fragment helpers
union FragU { v16bf f; v8bf h[2]; };

// two 16B chunks at p0 and p1 (global or LDS; compiler picks b128 op)
__device__ inline v16bf load2x16B(const bf16* p0, const bf16* p1) {
  FragU u;
  u.h[0] = *reinterpret_cast<const v8bf*>(p0);
  u.h[1] = *reinterpret_cast<const v8bf*>(p1);
  return u.f;
}

__device__ inline v8f wmma_bf16(v16bf a, v16bf b, v8f c) {
  // D = A(16x32 bf16) * B(32x16 bf16) + C(16x16 f32)
  return __builtin_amdgcn_wmma_f32_16x16x32_bf16(
      /*neg_a=*/false, a, /*neg_b=*/false, b,
      /*c_mod=*/(short)0, c, /*reuse_a=*/false, /*reuse_b=*/false);
}

// gfx1250 async copy global -> LDS (ASYNCcnt-tracked, bypasses VGPRs)
__device__ inline void async_g2l_b128(void* lds_ptr, const void* gptr) {
  unsigned l = (unsigned)(uintptr_t)lds_ptr;   // low 32 bits of flat addr == LDS byte addr
  asm volatile("global_load_async_to_lds_b128 %0, %1, off"
               :: "v"(l), "v"(gptr) : "memory");
}

// ---------------------------------------------------------------- DPP16 row max
// C-layout stripes one matrix row across a 16-lane DPP row; rotate-reduce
// with row_ror 8/4/2/1 entirely in the VALU (no LDS pipe traffic).
template<int CTRL>
__device__ inline float dpp_ror(float x) {
  int r = __builtin_amdgcn_mov_dpp(__builtin_bit_cast(int, x), CTRL, 0xF, 0xF, true);
  return __builtin_bit_cast(float, r);
}
__device__ inline float rowmax16(float x) {
  x = fmaxf(x, dpp_ror<0x128>(x));   // row_ror:8
  x = fmaxf(x, dpp_ror<0x124>(x));   // row_ror:4
  x = fmaxf(x, dpp_ror<0x122>(x));   // row_ror:2
  x = fmaxf(x, dpp_ror<0x121>(x));   // row_ror:1
  return x;
}

// ---------------------------------------------------------------- f32 -> bf16 converters
__global__ __launch_bounds__(256)
void cvt_f32_bf16(const float* __restrict__ src, bf16* __restrict__ dst, int n) {
  int i = (blockIdx.x * blockDim.x + threadIdx.x) * 4;
  if (i + 3 < n) {
    v4f v = *reinterpret_cast<const v4f*>(src + i);
    union { bf16 e[4]; unsigned long long u; } p;
    p.e[0] = (bf16)v.x; p.e[1] = (bf16)v.y; p.e[2] = (bf16)v.z; p.e[3] = (bf16)v.w;
    *reinterpret_cast<unsigned long long*>(dst + i) = p.u;
  }
}

// w[h][d][k] (f32) -> wt[h][k][d] (bf16): makes every WMMA B-column a contiguous row
__global__ __launch_bounds__(256)
void cvt_transpose_w(const float* __restrict__ src, bf16* __restrict__ dst, int n) {
  int i = blockIdx.x * blockDim.x + threadIdx.x;
  if (i >= n) return;
  int k = i & (DK - 1);
  int d = (i >> 6) & (DMODEL - 1);
  int h = i >> 16;
  dst[((size_t)h * DK + k) * DMODEL + d] = (bf16)src[i];
}

// ---------------------------------------------------------------- head projection GEMM
// X[BL][DMODEL] (bf16) x Wt[h][DK][DMODEL] -> Y (per head, M=BL, N=64, K=1024)
// TRANS_OUT=false : Y[h][BL][64]   (row-major, for Q/K fragment sources)
// TRANS_OUT=true  : Y[h][64][BL]   (feature-major, for V as P@V B-source)
template<bool TRANS_OUT>
__global__ __launch_bounds__(128)
void proj_gemm(const bf16* __restrict__ X, const bf16* __restrict__ Wt,
               bf16* __restrict__ Y) {
  const int h     = blockIdx.x >> 6;     // 16 heads
  const int mtile = blockIdx.x & 63;     // 64 tiles of 64 rows
  const int wave  = threadIdx.x >> 5;
  const int lane  = threadIdx.x & 31;
  const int m0    = mtile * 64 + wave * 16;

  const bf16* wt = Wt + (size_t)h * DK * DMODEL;

  v8f acc[4] = {};
  const int rowA  = m0 + (lane & 15);
  const int kaoff = (lane < 16) ? 0 : 16;   // A frag K half per half-wave
  const int kboff = (lane < 16) ? 0 : 16;   // B frag K half per half-wave

  for (int k0 = 0; k0 < DMODEL; k0 += 32) {
    const bf16* ap = X + (size_t)rowA * DMODEL + k0 + kaoff;
    v16bf a = load2x16B(ap, ap + 8);
#pragma unroll
    for (int c = 0; c < 4; ++c) {
      const bf16* bp = wt + (size_t)(c * 16 + (lane & 15)) * DMODEL + k0 + kboff;
      v16bf b = load2x16B(bp, bp + 8);
      acc[c] = wmma_bf16(a, b, acc[c]);
    }
  }

  if (!TRANS_OUT) {
    bf16* y = Y + ((size_t)h * BL + m0) * DK;
    const int rbase = (lane < 16) ? 0 : 8;
#pragma unroll
    for (int c = 0; c < 4; ++c) {
      int n = c * 16 + (lane & 15);
#pragma unroll
      for (int r = 0; r < 8; ++r)
        y[(size_t)(rbase + r) * DK + n] = (bf16)acc[c][r];
    }
  } else {
    const int mb = m0 + ((lane < 16) ? 0 : 8);
#pragma unroll
    for (int c = 0; c < 4; ++c) {
      int n = c * 16 + (lane & 15);
      v8bf pk;
#pragma unroll
      for (int r = 0; r < 8; ++r) pk[r] = (bf16)acc[c][r];
      *reinterpret_cast<v8bf*>(Y + ((size_t)h * DK + n) * BL + mb) = pk;
    }
  }
}

// ---------------------------------------------------------------- flash attention
// Qh/Kh: [h][BL][64] bf16, Vt: [h][64][BL] bf16 (feature-major)
// O: [BL][1024] bf16 with head concat on last dim; epilogue multiplies by qh.
// K/V tiles staged once per workgroup via async-to-LDS, double buffered.
// Row-sums for the softmax normalizer come from an extra WMMA against a
// ones-matrix (C-layout matches lrun[] register layout exactly).
#define LDS_STRIDE 48   // elements per P row (padded, keeps 16B alignment)

__global__ __launch_bounds__(128)
void flash_attn(const bf16* __restrict__ Qh, const bf16* __restrict__ Kh,
                const bf16* __restrict__ Vt, bf16* __restrict__ O) {
  __shared__ __align__(16) bf16 Kbuf[2][32][DK];        // [buf][key][feat]   8 KB
  __shared__ __align__(16) bf16 Vbuf[2][DK][32];        // [buf][feat][key]   8 KB
  __shared__ __align__(16) bf16 Pbuf[4][16][LDS_STRIDE];// per-wave P reshape 6 KB

  const int idx  = blockIdx.x;
  const int h    = idx >> 6;          // 16
  const int b    = (idx >> 5) & 1;    // 2
  const int lt   = idx & 31;          // 32 l-tiles of 64 rows
  const int tid  = threadIdx.x;
  const int wave = tid >> 5;
  const int lane = tid & 31;
  const int m0   = lt * 64 + wave * 16;   // query row within (h,b)

  const size_t hb = (size_t)h * BL + (size_t)b * SEQ;
  const bf16* q  = Qh + (hb + m0) * DK;
  const bf16* kk = Kh + hb * DK;
  const bf16* vt = Vt + (size_t)h * DK * BL + (size_t)b * SEQ;

  const int kaoff = (lane < 16) ? 0 : 16;
  const int kboff = (lane < 16) ? 0 : 16;
  const int rA    = lane & 15;

  // all-ones B fragment: row-sum WMMA operand
  v16bf onesf;
#pragma unroll
  for (int i = 0; i < 16; ++i) onesf[i] = (bf16)1.0f;

  // stage one 32-key K tile + V tile: 512 x 16B chunks, 4 async b128 per thread
  auto stage = [&](int j0, int buf) {
#pragma unroll
    for (int i = 0; i < 2; ++i) {
      int c = tid + i * 128;                 // 0..255
      int key = c >> 3, f8 = (c & 7) * 8;
      async_g2l_b128(&Kbuf[buf][key][f8], kk + (size_t)(j0 + key) * DK + f8);
      int feat = c >> 2, k8 = (c & 3) * 8;
      async_g2l_b128(&Vbuf[buf][feat][k8], vt + (size_t)feat * BL + j0 + k8);
    }
  };

  // Q A-fragments: 16x64 = two 16x32 fragments, kept in registers all loop
  const bf16* qrow = q + (size_t)rA * DK;
  v16bf qa0 = load2x16B(qrow + kaoff,      qrow + kaoff + 8);
  v16bf qa1 = load2x16B(qrow + 32 + kaoff, qrow + 32 + kaoff + 8);

  v8f o[4] = {};
  float mrun[8], lrun[8], alpha[8];
#pragma unroll
  for (int r = 0; r < 8; ++r) { mrun[r] = -1e30f; lrun[r] = 0.0f; }

  stage(0, 0);                                   // prime buffer 0

  for (int j0 = 0; j0 < SEQ; j0 += 32) {
    const int buf = (j0 >> 5) & 1;
    if (j0 + 32 < SEQ) {
      stage(j0 + 32, buf ^ 1);                   // prefetch next tile
      asm volatile("s_wait_asynccnt 0x4" ::: "memory");  // current tile landed
    } else {
      asm volatile("s_wait_asynccnt 0x0" ::: "memory");
    }
    __syncthreads();                             // tile visible to all waves

    // ---- S = Q @ K^T : two 16x16 f32 tiles over 32 keys (B frags from LDS)
    const bf16* kr0 = &Kbuf[buf][rA][kboff];
    const bf16* kr1 = &Kbuf[buf][16 + rA][kboff];
    v16bf b00 = load2x16B(kr0,      kr0 + 8);
    v16bf b01 = load2x16B(kr0 + 32, kr0 + 40);
    v16bf b10 = load2x16B(kr1,      kr1 + 8);
    v16bf b11 = load2x16B(kr1 + 32, kr1 + 40);
    v8f s0 = {}, s1 = {};
    s0 = wmma_bf16(qa0, b00, s0);
    s0 = wmma_bf16(qa1, b01, s0);
    s1 = wmma_bf16(qa0, b10, s1);
    s1 = wmma_bf16(qa1, b11, s1);

    // ---- online softmax: DPP rotate-reduce row max, exponentials, rescale
#pragma unroll
    for (int r = 0; r < 8; ++r) {
      float mx    = rowmax16(fmaxf(s0[r], s1[r]));
      float mnew  = fmaxf(mrun[r], mx);
      alpha[r]    = __expf(mrun[r] - mnew);
      mrun[r]     = mnew;
      s0[r] = __expf(s0[r] - mnew);
      s1[r] = __expf(s1[r] - mnew);
#pragma unroll
      for (int c = 0; c < 4; ++c) o[c][r] *= alpha[r];
    }

    // ---- reshape P: C-layout -> A-layout through per-wave LDS region
#pragma unroll
    for (int r = 0; r < 8; ++r) {
      int row = r + ((lane < 16) ? 0 : 8);
      Pbuf[wave][row][lane & 15]        = (bf16)s0[r];
      Pbuf[wave][row][16 + (lane & 15)] = (bf16)s1[r];
    }
    asm volatile("s_wait_dscnt 0x0" ::: "memory");
    FragU pu;
    pu.h[0] = *reinterpret_cast<const v8bf*>(&Pbuf[wave][rA][kaoff]);
    pu.h[1] = *reinterpret_cast<const v8bf*>(&Pbuf[wave][rA][kaoff + 8]);
    asm volatile("" ::: "memory");

    // ---- row sums on the matrix pipe: P @ ones -> every lane gets its row sum
    v8f zero = {};
    v8f rsv = wmma_bf16(pu.f, onesf, zero);
#pragma unroll
    for (int r = 0; r < 8; ++r)
      lrun[r] = lrun[r] * alpha[r] + rsv[r];

    // ---- O += P(16x32) @ V(32x64): V feature rows from LDS, keys contiguous
#pragma unroll
    for (int c = 0; c < 4; ++c) {
      const bf16* vp = &Vbuf[buf][c * 16 + rA][kboff];
      v16bf bv = load2x16B(vp, vp + 8);
      o[c] = wmma_bf16(pu.f, bv, o[c]);
    }
    __syncthreads();                   // everyone done reading buf before overwrite
  }

  // ---- epilogue: 1/l (fast rcp), * qh, head-concat store
#pragma unroll
  for (int r = 0; r < 8; ++r) {
    float inv = __builtin_amdgcn_rcpf(lrun[r]);
    int row = r + ((lane < 16) ? 0 : 8);
    const bf16* qr = q + (size_t)row * DK;
    size_t obase = ((size_t)b * SEQ + m0 + row) * DMODEL + (size_t)h * DK;
#pragma unroll
    for (int c = 0; c < 4; ++c) {
      int n = c * 16 + (lane & 15);
      O[obase + n] = (bf16)(o[c][r] * inv * (float)qr[n]);
    }
  }
}

// ---------------------------------------------------------------- output projection
// X[BL][1024] bf16 x w_proj[1024][1024] (B column n == w_proj row n) + bias -> f32
__global__ __launch_bounds__(128)
void out_proj(const bf16* __restrict__ X, const bf16* __restrict__ W,
              const float* __restrict__ bias, float* __restrict__ Y) {
  const int mtile = blockIdx.x >> 4;   // 64
  const int ntile = blockIdx.x & 15;   // 16
  const int wave  = threadIdx.x >> 5;
  const int lane  = threadIdx.x & 31;
  const int m0    = mtile * 64 + wave * 16;
  const int n0    = ntile * 64;

  v8f acc[4] = {};
  const int rowA  = m0 + (lane & 15);
  const int kaoff = (lane < 16) ? 0 : 16;
  const int kboff = (lane < 16) ? 0 : 16;

  for (int k0 = 0; k0 < DMODEL; k0 += 32) {
    const bf16* ap = X + (size_t)rowA * DMODEL + k0 + kaoff;
    v16bf a = load2x16B(ap, ap + 8);
#pragma unroll
    for (int c = 0; c < 4; ++c) {
      const bf16* bp = W + (size_t)(n0 + c * 16 + (lane & 15)) * DMODEL + k0 + kboff;
      v16bf bw = load2x16B(bp, bp + 8);
      acc[c] = wmma_bf16(a, bw, acc[c]);
    }
  }

  const int rbase = (lane < 16) ? 0 : 8;
#pragma unroll
  for (int c = 0; c < 4; ++c) {
    int n = n0 + c * 16 + (lane & 15);
    float bv = bias[n];
#pragma unroll
    for (int r = 0; r < 8; ++r)
      Y[(size_t)(m0 + rbase + r) * DMODEL + n] = acc[c][r] + bv;
  }
}

// ---------------------------------------------------------------- host launcher
extern "C" void kernel_launch(void* const* d_in, const int* in_sizes, int n_in,
                              void* d_out, int out_size, void* d_ws, size_t ws_size,
                              hipStream_t stream) {
  const float* q      = (const float*)d_in[0];
  const float* k      = (const float*)d_in[1];
  const float* v      = (const float*)d_in[2];
  const float* w_qs   = (const float*)d_in[3];
  const float* w_ks   = (const float*)d_in[4];
  const float* w_vs   = (const float*)d_in[5];
  const float* w_proj = (const float*)d_in[6];
  const float* b_proj = (const float*)d_in[7];
  float* out = (float*)d_out;

  char* ws = (char*)d_ws;
  size_t off = 0;
  auto alloc = [&](size_t bytes) -> void* {
    void* p = ws + off;
    off += (bytes + 255) & ~(size_t)255;
    return p;
  };

  const int NQ = BL * DMODEL;           // 4,194,304
  const int NW = NHEADS * DMODEL * DK;  // 1,048,576
  const int NP = DMODEL * DMODEL;       // 1,048,576
  const int NH = NHEADS * BL * DK;      // 4,194,304

  bf16* qb  = (bf16*)alloc((size_t)NQ * 2);
  bf16* kb  = (bf16*)alloc((size_t)NQ * 2);
  bf16* vb  = (bf16*)alloc((size_t)NQ * 2);
  bf16* wtq = (bf16*)alloc((size_t)NW * 2);
  bf16* wtk = (bf16*)alloc((size_t)NW * 2);
  bf16* wtv = (bf16*)alloc((size_t)NW * 2);
  bf16* wpb = (bf16*)alloc((size_t)NP * 2);
  bf16* qh  = (bf16*)alloc((size_t)NH * 2);
  bf16* kh  = (bf16*)alloc((size_t)NH * 2);
  bf16* vht = (bf16*)alloc((size_t)NH * 2);
  bf16* ao  = (bf16*)alloc((size_t)NQ * 2);

  // precision downconvert + weight transposition
  cvt_f32_bf16<<<NQ / 4 / 256, 256, 0, stream>>>(q, qb, NQ);
  cvt_f32_bf16<<<NQ / 4 / 256, 256, 0, stream>>>(k, kb, NQ);
  cvt_f32_bf16<<<NQ / 4 / 256, 256, 0, stream>>>(v, vb, NQ);
  cvt_f32_bf16<<<NP / 4 / 256, 256, 0, stream>>>(w_proj, wpb, NP);
  cvt_transpose_w<<<NW / 256, 256, 0, stream>>>(w_qs, wtq, NW);
  cvt_transpose_w<<<NW / 256, 256, 0, stream>>>(w_ks, wtk, NW);
  cvt_transpose_w<<<NW / 256, 256, 0, stream>>>(w_vs, wtv, NW);

  // per-head projections (V written feature-major for P@V fragment loads)
  proj_gemm<false><<<NHEADS * 64, 128, 0, stream>>>(qb, wtq, qh);
  proj_gemm<false><<<NHEADS * 64, 128, 0, stream>>>(kb, wtk, kh);
  proj_gemm<true ><<<NHEADS * 64, 128, 0, stream>>>(vb, wtv, vht);

  // fused attention: async-staged K/V tiles, online softmax, *qh epilogue
  flash_attn<<<NHEADS * BATCH * (SEQ / 64), 128, 0, stream>>>(qh, kh, vht, ao);

  // final projection + bias, f32 output
  out_proj<<<(BL / 64) * (DMODEL / 64), 128, 0, stream>>>(ao, wpb, b_proj, out);
}
